// MetaLearner_33621003993399
// MI455X (gfx1250) — matrix operand order repaired and checked by reading, not compiled, not run on hardware
//
#include <hip/hip_runtime.h>
#include <math.h>

#define BB 8
#define TT 6
#define NN 64
#define DD 512
#define CC 128
#define UU 20
#define DC (DD*CC)          // 65536
#define PP (DC + CC)        // 65664
#define BP ((size_t)BB*PP)  // 525312

typedef __attribute__((ext_vector_type(16))) _Float16 v16h;
typedef __attribute__((ext_vector_type(8)))  float    v8f;

__device__ __forceinline__ float sigmoidf_(float x) { return 1.0f / (1.0f + expf(-x)); }

// _preprocess from the reference (Pc = 10)
__device__ __forceinline__ void preprocess_(float x, float& a, float& b) {
    const float Pc = 10.0f;
    const float eP  = expf(Pc);
    const float enP = expf(-Pc);
    float ax = fabsf(x);
    if (ax > enP) {               // m1 branch
        a = logf(ax) / Pc;
        b = (x > 0.0f) ? 1.0f : ((x < 0.0f) ? -1.0f : 0.0f);
    } else {                      // m2 branch
        a = -1.0f;
        b = eP * x;
    }
}

// ---------------------------------------------------------------------------
// Step 1: preds = feats_t @ W + bias ; dpreds = 2*(preds-labels)/C ; loss[b] += sum diff^2 / C
// One wave (32 threads) per 16x16 output tile. Grid = B * (N/16) * (C/16) = 256.
// ---------------------------------------------------------------------------
__global__ __launch_bounds__(32) void fwd_kernel(const float* __restrict__ feats,
                                                 const float* __restrict__ labels,
                                                 const float* __restrict__ prm,
                                                 float* __restrict__ dpreds,
                                                 float* __restrict__ loss,
                                                 int t)
{
    int wid  = blockIdx.x;
    int b    = wid / 32;          // 32 tiles per batch
    int tile = wid % 32;
    int m0 = (tile / 8) * 16;     // n-tile
    int n0 = (tile % 8) * 16;     // c-tile
    int lane = threadIdx.x;
    int l  = lane & 15;
    int hi = (lane >= 16) ? 1 : 0;

    const float* A    = feats  + ((size_t)b*TT + t) * NN * DD;   // (N,D) row-major
    const float* W    = prm    + (size_t)b * PP;                 // (D,C) row-major
    const float* bias = W + DC;
    const float* L    = labels + ((size_t)b*TT + t) * NN * CC;

    v8f acc = {};
    for (int k0 = 0; k0 < DD; k0 += 32) {
        v16h a, bm;
        // A fragment: 16B-aligned contiguous runs -> b128 loads
        const float4* ar4 = reinterpret_cast<const float4*>(A + (size_t)(m0 + l) * DD + k0 + (hi ? 8 : 0));
        float4 c0 = ar4[0], c1 = ar4[1];   // K = base + 0..7
        float4 c2 = ar4[4], c3 = ar4[5];   // K = base + 16..23
        a[0]  = (_Float16)c0.x; a[1]  = (_Float16)c0.y; a[2]  = (_Float16)c0.z; a[3]  = (_Float16)c0.w;
        a[4]  = (_Float16)c1.x; a[5]  = (_Float16)c1.y; a[6]  = (_Float16)c1.z; a[7]  = (_Float16)c1.w;
        a[8]  = (_Float16)c2.x; a[9]  = (_Float16)c2.y; a[10] = (_Float16)c2.z; a[11] = (_Float16)c2.w;
        a[12] = (_Float16)c3.x; a[13] = (_Float16)c3.y; a[14] = (_Float16)c3.z; a[15] = (_Float16)c3.w;
        const float* br = W + (size_t)(k0 + (hi ? 8 : 0)) * CC + n0 + l;
        #pragma unroll
        for (int j = 0; j < 16; ++j) bm[j] = (_Float16)br[(size_t)(j + (j & 8)) * CC];
        acc = __builtin_amdgcn_wmma_f32_16x16x32_f16(false, a, false, bm, (short)0, acc, false, false);
    }

    float bv   = bias[n0 + l];
    float part = 0.0f;
    #pragma unroll
    for (int r = 0; r < 8; ++r) {
        int m = m0 + r + (hi ? 8 : 0);
        int c = n0 + l;
        float pred = acc[r] + bv;
        float diff = pred - L[(size_t)m * CC + c];
        part += diff * diff;
        dpreds[((size_t)b * NN + m) * CC + c] = 2.0f * diff / (float)CC;
    }
    #pragma unroll
    for (int off = 16; off > 0; off >>= 1)
        part += __shfl_xor(part, off, 32);
    if (lane == 0) atomicAdd(&loss[b], part / (float)CC);
}

// ---------------------------------------------------------------------------
// Step 2: gradW[b] = feats_t^T @ dpreds  (512x64 @ 64x128), gradB = colsum(dpreds)
// One wave per 16x16 tile; K=64 -> 2 WMMAs. Grid = B*256 (+B bias blocks).
// ---------------------------------------------------------------------------
__global__ __launch_bounds__(32) void bwd_kernel(const float* __restrict__ feats,
                                                 const float* __restrict__ dpreds,
                                                 float* __restrict__ grads,
                                                 int t)
{
    int wid = blockIdx.x;
    if (wid >= BB * 256) {        // bias-gradient blocks
        int b = wid - BB * 256;
        int lane = threadIdx.x;
        const float* dp = dpreds + (size_t)b * NN * CC;
        for (int c = lane; c < CC; c += 32) {
            float s = 0.0f;
            for (int n = 0; n < NN; ++n) s += dp[(size_t)n * CC + c];
            grads[(size_t)b * PP + DC + c] = s;
        }
        return;
    }
    int b    = wid / 256;
    int tile = wid % 256;         // (D/16)*(C/16) = 32*8
    int m0 = (tile / 8) * 16;     // d-tile
    int n0 = (tile % 8) * 16;     // c-tile
    int lane = threadIdx.x;
    int l  = lane & 15;
    int hi = (lane >= 16) ? 1 : 0;

    const float* F  = feats  + ((size_t)b*TT + t) * NN * DD;   // (N,D); A = F^T (D,N)
    const float* dp = dpreds + (size_t)b * NN * CC;

    v8f acc = {};
    for (int k0 = 0; k0 < NN; k0 += 32) {
        v16h a, bm;
        int kb = k0 + (hi ? 8 : 0);
        #pragma unroll
        for (int j = 0; j < 8; ++j) {
            a[j]     = (_Float16)F[(size_t)(kb + j)      * DD + m0 + l];
            a[8 + j] = (_Float16)F[(size_t)(kb + 16 + j) * DD + m0 + l];
        }
        int kbb = k0 + (hi ? 8 : 0);
        #pragma unroll
        for (int j = 0; j < 16; ++j) bm[j] = (_Float16)dp[(size_t)(kbb + j + (j & 8)) * CC + n0 + l];
        acc = __builtin_amdgcn_wmma_f32_16x16x32_f16(false, a, false, bm, (short)0, acc, false, false);
    }
    #pragma unroll
    for (int r = 0; r < 8; ++r) {
        int d = m0 + r + (hi ? 8 : 0);
        int c = n0 + l;
        grads[(size_t)b * PP + (size_t)d * CC + c] = acc[r];
    }
}

// ---------------------------------------------------------------------------
// Step 3: fused coordinate-wise LSTM over all B*P rows.
// Wave handles 16 rows; z (16x80) = [mli|h|0] (16x32) @ Wcomb (32x80) in 5 WMMAs.
// Then elementwise cell update + f/i gates + param update, one global pass.
// Grid = 4104 blocks * 256 threads (8 waves) = BP/16 waves exactly.
// ---------------------------------------------------------------------------
__global__ __launch_bounds__(256) void lstm_kernel(const float* __restrict__ prm_in,
                                                   const float* __restrict__ grads,
                                                   const float* __restrict__ loss,
                                                   float* __restrict__ h,
                                                   float* __restrict__ cst,
                                                   float* __restrict__ fprev,
                                                   float* __restrict__ iprev,
                                                   const float* __restrict__ k5,    // (5,80)
                                                   const float* __restrict__ rec,   // (20,80)
                                                   const float* __restrict__ zb,    // (80)
                                                   const float* __restrict__ Wf,    // (21)
                                                   const float* __restrict__ bf,    // (1)
                                                   const float* __restrict__ Wi,    // (21)
                                                   const float* __restrict__ bi,    // (1)
                                                   float* __restrict__ prm_out)
{
    __shared__ float zlds[8][16][80];   // 40 KB
    __shared__ float hlds[8][16][20];   // 10 KB

    int wave = threadIdx.x >> 5;
    int lane = threadIdx.x & 31;
    int l  = lane & 15;
    int hi = (lane >= 16) ? 1 : 0;
    size_t tile = (size_t)blockIdx.x * 8 + wave;
    size_t r0   = tile * 16;

    // --- B fragments: Wcomb rows 0-4 = lstm_kernel, 5-24 = lstm_rec, 25-31 = 0
    v16h bfrag[5];
    #pragma unroll
    for (int ct = 0; ct < 5; ++ct) {
        int n = ct * 16 + l;
        #pragma unroll
        for (int j = 0; j < 16; ++j) {
            int krow = (hi ? 8 : 0) + j + (j & 8);
            float w;
            if      (krow < 5)  w = k5[krow * 80 + n];
            else if (krow < 25) w = rec[(krow - 5) * 80 + n];
            else                w = 0.0f;
            bfrag[ct][j] = (_Float16)w;
        }
    }

    // --- A fragment: columns 0-4 = [prm,g0,g1,l0,l1], 5-24 = h[0..19], 25-31 = 0
    size_t r = r0 + l;                       // lanes L and L+16 share row r
    int bidx = (int)(r / PP);
    float prmv = prm_in[r];
    float gv   = grads[r];
    float lv   = loss[bidx];
    float g0, g1, l0v, l1v;
    preprocess_(gv, g0, g1);
    preprocess_(lv, l0v, l1v);

    // h row: 20 floats = five 16B-aligned float4s -> b128 loads
    float hv[UU];
    {
        const float4* h4 = reinterpret_cast<const float4*>(h + r * UU);
        #pragma unroll
        for (int q = 0; q < 5; ++q) {
            float4 v = h4[q];
            hv[4*q + 0] = v.x; hv[4*q + 1] = v.y; hv[4*q + 2] = v.z; hv[4*q + 3] = v.w;
        }
    }

    v16h afrag;
    if (!hi) {
        afrag[0] = (_Float16)prmv; afrag[1] = (_Float16)g0;  afrag[2] = (_Float16)g1;
        afrag[3] = (_Float16)l0v;  afrag[4] = (_Float16)l1v;
        afrag[5] = (_Float16)hv[0]; afrag[6] = (_Float16)hv[1]; afrag[7] = (_Float16)hv[2];
        #pragma unroll
        for (int j = 0; j < 8; ++j) afrag[8 + j] = (_Float16)hv[11 + j];   // cols 16..23
    } else {
        #pragma unroll
        for (int j = 0; j < 8; ++j) afrag[j] = (_Float16)hv[3 + j];        // cols 8..15
        afrag[8] = (_Float16)hv[19];                                       // col 24
        #pragma unroll
        for (int j = 1; j < 8; ++j) afrag[8 + j] = (_Float16)0.0f;         // cols 25..31
    }

    // --- z = A @ Wcomb : one WMMA per 16-column tile
    v8f z[5];
    #pragma unroll
    for (int ct = 0; ct < 5; ++ct) {
        v8f zero = {};
        z[ct] = __builtin_amdgcn_wmma_f32_16x16x32_f16(false, afrag, false, bfrag[ct],
                                                       (short)0, zero, false, false);
    }

    // --- + bias, stash to LDS in row-major
    #pragma unroll
    for (int ct = 0; ct < 5; ++ct) {
        float bv = zb[ct * 16 + l];
        #pragma unroll
        for (int rr = 0; rr < 8; ++rr)
            zlds[wave][rr + (hi ? 8 : 0)][ct * 16 + l] = z[ct][rr] + bv;
    }
    __syncthreads();

    // --- LSTM cell: tile state is 16*20 = 320 contiguous floats = 80 float4s.
    //     float4 granularity (U divisible by 4 -> never straddles a row).
    {
        float4*       c4 = reinterpret_cast<float4*>(cst + r0 * UU);
        float4*       h4 = reinterpret_cast<float4*>(h   + r0 * UU);
        for (int it4 = lane; it4 < 80; it4 += 32) {
            int f0 = it4 * 4;
            int mm = f0 / UU;
            int u0 = f0 % UU;
            float4 cold = c4[it4];
            float co[4] = { cold.x, cold.y, cold.z, cold.w };
            float cn[4], hn[4];
            #pragma unroll
            for (int j = 0; j < 4; ++j) {
                int u = u0 + j;
                float zi = zlds[wave][mm][u];
                float zf = zlds[wave][mm][20 + u];
                float zc = zlds[wave][mm][40 + u];
                float zo = zlds[wave][mm][60 + u];
                cn[j] = sigmoidf_(zf) * co[j] + sigmoidf_(zi) * tanhf(zc);
                hn[j] = sigmoidf_(zo) * tanhf(cn[j]);
            }
            float4 cnv = make_float4(cn[0], cn[1], cn[2], cn[3]);
            float4 hnv = make_float4(hn[0], hn[1], hn[2], hn[3]);
            c4[it4] = cnv;
            h4[it4] = hnv;
            *reinterpret_cast<float4*>(&hlds[wave][mm][u0]) = hnv;   // ds_store_b128
        }
    }
    __syncthreads();

    // --- gates f, i and param update: one row per low lane
    if (!hi) {
        size_t rr = r0 + l;
        float accf = bf[0] + fprev[rr] * Wf[20];
        float acci = bi[0] + iprev[rr] * Wi[20];
        #pragma unroll
        for (int u = 0; u < UU; ++u) {
            float hvv = hlds[wave][l][u];
            accf += hvv * Wf[u];
            acci += hvv * Wi[u];
        }
        float f  = sigmoidf_(accf);
        float ii = sigmoidf_(acci);
        prm_out[rr] = f * prmv - ii * gv;
        fprev[rr] = f;
        iprev[rr] = ii;
    }
}

// ---------------------------------------------------------------------------
__global__ void init_kernel(const float* __restrict__ params,
                            float* __restrict__ h, float* __restrict__ c,
                            float* __restrict__ fp, float* __restrict__ ip,
                            float* __restrict__ prmA)
{
    size_t i = (size_t)blockIdx.x * blockDim.x + threadIdx.x;
    size_t stride = (size_t)gridDim.x * blockDim.x;
    for (size_t k = i; k < BP * UU; k += stride) { h[k] = 0.0f; c[k] = 0.0f; }
    for (size_t k = i; k < BP; k += stride) { fp[k] = 0.0f; ip[k] = 0.0f; prmA[k] = params[k]; }
}

extern "C" void kernel_launch(void* const* d_in, const int* in_sizes, int n_in,
                              void* d_out, int out_size, void* d_ws, size_t ws_size,
                              hipStream_t stream)
{
    const float* feats  = (const float*)d_in[0];
    const float* labels = (const float*)d_in[1];
    const float* params = (const float*)d_in[2];
    const float* k5     = (const float*)d_in[3];
    const float* rec    = (const float*)d_in[4];
    const float* zb     = (const float*)d_in[5];
    const float* Wf     = (const float*)d_in[6];
    const float* bf     = (const float*)d_in[7];
    const float* Wi     = (const float*)d_in[8];
    const float* bi     = (const float*)d_in[9];
    float* out = (float*)d_out;

    float* ws = (float*)d_ws;
    size_t off = 0;
    float* h      = ws + off; off += BP * UU;
    float* c      = ws + off; off += BP * UU;
    float* prmA   = ws + off; off += BP;
    float* prmB   = ws + off; off += BP;
    float* fp     = ws + off; off += BP;
    float* ip     = ws + off; off += BP;
    float* grads  = ws + off; off += BP;
    float* dpreds = ws + off; off += (size_t)BB * NN * CC;
    float* loss   = ws + off; off += BB;

    init_kernel<<<2048, 256, 0, stream>>>(params, h, c, fp, ip, prmA);

    float* pin  = prmA;
    float* pout = prmB;
    for (int t = 0; t < TT; ++t) {
        hipMemsetAsync(loss, 0, BB * sizeof(float), stream);
        fwd_kernel<<<256, 32, 0, stream>>>(feats, labels, pin, dpreds, loss, t);
        bwd_kernel<<<BB * 256 + BB, 32, 0, stream>>>(feats, dpreds, grads, t);
        float* dst = (t == TT - 1) ? out : pout;
        lstm_kernel<<<4104, 256, 0, stream>>>(pin, grads, loss, h, c, fp, ip,
                                              k5, rec, zb, Wf, bf, Wi, bi, dst);
        float* tmp = pin; pin = pout; pout = tmp;
    }
}
